// GAT_33071248179267
// MI455X (gfx1250) — matrix operand order repaired and checked by reading, not compiled, not run on hardware
//
#include <hip/hip_runtime.h>
#include <hip/hip_bf16.h>
#include <cstdint>

// ---- problem constants (match reference) ----
#define NPG     186      // nodes per graph
#define FIN     5
#define HEADS   4
#define COUT    24
#define KW      62
#define OCH     8
#define TCONV   125      // 186 - 62 + 1
#define KLIN    1488     // COUT*KW (im2col K)
#define KPAD    1504     // 47 * 32 (K padded to WMMA k-step)
#define GELEMS  4464     // NPG*COUT floats per graph of GAT output

typedef __attribute__((ext_vector_type(16))) _Float16 v16h;
typedef __attribute__((ext_vector_type(8)))  _Float16 v8h;
typedef __attribute__((ext_vector_type(8)))  float    v8f;

// Native CDNA5 non-returning FP32 global atomic add, device scope.
// Guarantees GLOBAL_ATOMIC_ADD_F32 (STOREcnt-tracked) instead of a CAS loop.
__device__ __forceinline__ void atomAddF(float* p, float v) {
  asm volatile("global_atomic_add_f32 %0, %1, off scope:SCOPE_DEV"
               :: "v"(p), "v"(v) : "memory");
}
__device__ __forceinline__ void atomAddD(double* p, double v) {
  __hip_atomic_fetch_add(p, v, __ATOMIC_RELAXED, __HIP_MEMORY_SCOPE_AGENT);
}
__device__ __forceinline__ void atomAddDWg(double* p, double v) {
  __hip_atomic_fetch_add(p, v, __ATOMIC_RELAXED, __HIP_MEMORY_SCOPE_WORKGROUP);
}

// ---------------- zero fill ----------------
__global__ void k_zero(float* __restrict__ p, long long n) {
  long long i  = (long long)blockIdx.x * blockDim.x + threadIdx.x;
  long long st = (long long)gridDim.x * blockDim.x;
  for (; i < n; i += st) p[i] = 0.0f;
}

// ---------------- BatchNorm statistics (sum, sumsq per feature) ----------------
__global__ void k_bnstats(const float* __restrict__ x, double* __restrict__ sums, int N) {
  __shared__ double s[2 * FIN];
  if (threadIdx.x < 2 * FIN) s[threadIdx.x] = 0.0;
  __syncthreads();
  double ls[FIN] = {0, 0, 0, 0, 0};
  double lq[FIN] = {0, 0, 0, 0, 0};
  long long i  = (long long)blockIdx.x * blockDim.x + threadIdx.x;
  long long st = (long long)gridDim.x * blockDim.x;
  for (long long n = i; n < N; n += st) {
#pragma unroll
    for (int f = 0; f < FIN; ++f) {
      float v = x[n * FIN + f];
      ls[f] += (double)v;
      lq[f] += (double)v * (double)v;
    }
  }
#pragma unroll
  for (int f = 0; f < FIN; ++f) {
    atomAddDWg(&s[f], ls[f]);
    atomAddDWg(&s[FIN + f], lq[f]);
  }
  __syncthreads();
  if (threadIdx.x < 2 * FIN) atomAddD(&sums[threadIdx.x], s[threadIdx.x]);
}

// ------- BN-apply + linear projection + attention coefficients (per node,head) -------
__global__ void k_linear(const float* __restrict__ x, const double* __restrict__ sums,
                         const float* __restrict__ gamma, const float* __restrict__ beta,
                         const float* __restrict__ lin_w, const float* __restrict__ att_src,
                         const float* __restrict__ att_dst,
                         float* __restrict__ h, float* __restrict__ a_s, float* __restrict__ a_d,
                         int N) {
  int tid = blockIdx.x * blockDim.x + threadIdx.x;
  if (tid >= N * HEADS) return;
  int n = tid >> 2, head = tid & 3;
  float xn[FIN];
#pragma unroll
  for (int f = 0; f < FIN; ++f) {
    double mu  = sums[f] / (double)N;
    double var = sums[FIN + f] / (double)N - mu * mu;
    float  sc  = rsqrtf((float)var + 1e-5f) * gamma[f];
    xn[f] = (x[(size_t)n * FIN + f] - (float)mu) * sc + beta[f];
  }
  const float* W  = lin_w + head * COUT * FIN;
  const float* As = att_src + head * COUT;
  const float* Ad = att_dst + head * COUT;
  float* hout = h + (size_t)n * (HEADS * COUT) + head * COUT;
  float as_ = 0.0f, ad_ = 0.0f;
#pragma unroll
  for (int c = 0; c < COUT; ++c) {
    const float* w = W + c * FIN;
    float v = xn[0]*w[0] + xn[1]*w[1] + xn[2]*w[2] + xn[3]*w[3] + xn[4]*w[4];
    hout[c] = v;
    as_ += v * As[c];
    ad_ += v * Ad[c];
  }
  a_s[tid] = as_;
  a_d[tid] = ad_;
}

// ---------------- softmax denominator over edges (+ self loops) ----------------
__global__ void k_edge_denom(const int* __restrict__ ei,
                             const float* __restrict__ a_s, const float* __restrict__ a_d,
                             float* __restrict__ denom, int E, int N) {
  int i = blockIdx.x * blockDim.x + threadIdx.x;
  if (i >= E + N) return;
  int s = (i < E) ? ei[i]     : (i - E);
  int d = (i < E) ? ei[E + i] : (i - E);
  float4 as4 = *(const float4*)(a_s + 4 * (size_t)s);
  float4 ad4 = *(const float4*)(a_d + 4 * (size_t)d);
  float l0 = as4.x + ad4.x, l1 = as4.y + ad4.y, l2 = as4.z + ad4.z, l3 = as4.w + ad4.w;
  l0 = (l0 > 0.f) ? l0 : 0.2f * l0;  l1 = (l1 > 0.f) ? l1 : 0.2f * l1;
  l2 = (l2 > 0.f) ? l2 : 0.2f * l2;  l3 = (l3 > 0.f) ? l3 : 0.2f * l3;
  float* dn = denom + 4 * (size_t)d;
  atomAddF(dn + 0, expf(l0));
  atomAddF(dn + 1, expf(l1));
  atomAddF(dn + 2, expf(l2));
  atomAddF(dn + 3, expf(l3));
}

// ---------------- attention-weighted aggregation (head mean folded in) ----------------
__global__ void k_edge_aggr(const int* __restrict__ ei,
                            const float* __restrict__ a_s, const float* __restrict__ a_d,
                            const float* __restrict__ denom, const float* __restrict__ h,
                            float* __restrict__ accum, int E, int N) {
  int i = blockIdx.x * blockDim.x + threadIdx.x;
  if (i >= E + N) return;
  int s = (i < E) ? ei[i]     : (i - E);
  int d = (i < E) ? ei[E + i] : (i - E);
  float4 as4 = *(const float4*)(a_s + 4 * (size_t)s);
  float4 ad4 = *(const float4*)(a_d + 4 * (size_t)d);
  float4 dn4 = *(const float4*)(denom + 4 * (size_t)d);
  float l0 = as4.x + ad4.x, l1 = as4.y + ad4.y, l2 = as4.z + ad4.z, l3 = as4.w + ad4.w;
  l0 = (l0 > 0.f) ? l0 : 0.2f * l0;  l1 = (l1 > 0.f) ? l1 : 0.2f * l1;
  l2 = (l2 > 0.f) ? l2 : 0.2f * l2;  l3 = (l3 > 0.f) ? l3 : 0.2f * l3;
  // 0.25 = mean over heads, folded into attn
  float at0 = 0.25f * expf(l0) / (dn4.x + 1e-16f);
  float at1 = 0.25f * expf(l1) / (dn4.y + 1e-16f);
  float at2 = 0.25f * expf(l2) / (dn4.z + 1e-16f);
  float at3 = 0.25f * expf(l3) / (dn4.w + 1e-16f);
  const float* hs = h + (size_t)s * (HEADS * COUT);
  float* acc = accum + (size_t)d * COUT;
#pragma unroll
  for (int c4 = 0; c4 < COUT / 4; ++c4) {
    float4 h0 = *(const float4*)(hs + 0 * COUT + c4 * 4);
    float4 h1 = *(const float4*)(hs + 1 * COUT + c4 * 4);
    float4 h2 = *(const float4*)(hs + 2 * COUT + c4 * 4);
    float4 h3 = *(const float4*)(hs + 3 * COUT + c4 * 4);
    atomAddF(acc + c4 * 4 + 0, h0.x*at0 + h1.x*at1 + h2.x*at2 + h3.x*at3);
    atomAddF(acc + c4 * 4 + 1, h0.y*at0 + h1.y*at1 + h2.y*at2 + h3.y*at3);
    atomAddF(acc + c4 * 4 + 2, h0.z*at0 + h1.z*at1 + h2.z*at2 + h3.z*at3);
    atomAddF(acc + c4 * 4 + 3, h0.w*at0 + h1.w*at1 + h2.w*at2 + h3.w*at3);
  }
}

// ---------------- bias + ELU, convert to f16 for the WMMA conv ----------------
__global__ void k_elu(const float* __restrict__ accum, const float* __restrict__ gbias,
                      _Float16* __restrict__ gatH, long long n) {
  long long i = (long long)blockIdx.x * blockDim.x + threadIdx.x;
  if (i >= n) return;
  float v = accum[i] + gbias[i % COUT];
  v = (v > 0.0f) ? v : expm1f(v);
  gatH[i] = (_Float16)v;
}

// -------- rearrange conv weights into f16 B-matrix Wh[o][k_lin], k_lin = k*24 + i --------
__global__ void k_prep_w(const float* __restrict__ conv_w, _Float16* __restrict__ Wh) {
  int tid = blockIdx.x * blockDim.x + threadIdx.x;
  if (tid >= 16 * KPAD) return;
  int o = tid / KPAD, kl = tid % KPAD;
  _Float16 v = (_Float16)0.0f;
  if (o < OCH && kl < KLIN) {
    int k = kl / COUT, i = kl % COUT;
    v = (_Float16)conv_w[(size_t)o * KLIN + (size_t)i * KW + k];
  }
  Wh[(size_t)o * KPAD + kl] = v;
}

// ---------------- Conv1d as batched GEMM via WMMA f32<=f16 16x16x32 ----------------
// Grid: B blocks x 256 threads (8 waves). Wave w computes the 16x16 output tile
// D[t = 16w .. 16w+15][o = 0..15] for graph b, K-loop over KPAD=1504 (47 wmma).
// A[t][k] = gatH[b*4464 + t*24 + k] (contiguous im2col window), B = Wh (zero-padded).
__global__ void __launch_bounds__(256) k_conv(const _Float16* __restrict__ gatH,
                                              const _Float16* __restrict__ Wh,
                                              const float* __restrict__ convb,
                                              float* __restrict__ out) {
  int b    = blockIdx.x;
  int wave = threadIdx.x >> 5;
  int lane = threadIdx.x & 31;
  int hs   = lane >> 4;      // half-wave select
  int l16  = lane & 15;

  const _Float16* G    = gatH + (size_t)b * GELEMS;
  const _Float16* Arow = G + (size_t)(wave * 16 + l16) * COUT;      // row M = l16 of this tile
  const _Float16* Brow = Wh + (size_t)l16 * KPAD + (hs ? 8 : 0);    // col N = l16

  v8f acc = {0.f, 0.f, 0.f, 0.f, 0.f, 0.f, 0.f, 0.f};
  for (int k0 = 0; k0 < KPAD; k0 += 32) {
    union { v16h v; v8h h[2]; } a, w;
    // A (16-bit 16x32): lanes 0-15 hold K {k0..k0+7, k0+16..k0+23}; lanes 16-31 are +8.
    a.h[0] = *(const v8h*)(Arow + k0 + (hs << 3));
    a.h[1] = *(const v8h*)(Arow + k0 + 16 + (hs << 3));
    // B (16-bit 32x16): lanes 0-15 hold K {k0..k0+7, k0+16..k0+23} of col l16; lanes 16-31 are +8.
    w.h[0] = *(const v8h*)(Brow + k0);
    w.h[1] = *(const v8h*)(Brow + k0 + 16);
    acc = __builtin_amdgcn_wmma_f32_16x16x32_f16(false, a.v, false, w.v,
                                                 (short)0, acc, false, false);
  }

  // D (32-bit 16x16): lane = col N; VGPR r holds row M = r + 8*hs.
  if (l16 < OCH) {
    float bo = convb[l16];
    float* outp = out + (size_t)b * (OCH * TCONV) + (size_t)l16 * TCONV;
#pragma unroll
    for (int r = 0; r < 8; ++r) {
      int t = wave * 16 + (hs << 3) + r;
      if (t < TCONV) {
        float y = acc[r] + bo;
        outp[t] = (y > 0.0f) ? y : 0.01f * y;
      }
    }
  }
}

// ---------------- host launcher ----------------
extern "C" void kernel_launch(void* const* d_in, const int* in_sizes, int n_in,
                              void* d_out, int out_size, void* d_ws, size_t ws_size,
                              hipStream_t stream) {
  const float* x       = (const float*)d_in[0];
  const int*   ei      = (const int*)d_in[1];
  // d_in[2] batch: unused (graph id = node / 186)
  const float* gamma   = (const float*)d_in[3];
  const float* beta    = (const float*)d_in[4];
  const float* lin_w   = (const float*)d_in[5];
  const float* att_src = (const float*)d_in[6];
  const float* att_dst = (const float*)d_in[7];
  const float* gbias   = (const float*)d_in[8];
  const float* conv_w  = (const float*)d_in[9];
  const float* conv_b  = (const float*)d_in[10];

  const int N = in_sizes[0] / FIN;
  const int E = in_sizes[1] / 2;
  const int B = N / NPG;

  // workspace layout (256B aligned regions)
  size_t off = 0;
  auto alloc = [&](size_t bytes) { size_t o = off; off = (off + bytes + 255) & ~(size_t)255; return o; };
  size_t o_sums  = alloc(2 * FIN * sizeof(double));
  size_t o_denom = alloc((size_t)N * HEADS * sizeof(float));
  size_t o_accum = alloc((size_t)N * COUT * sizeof(float));
  size_t zeroEnd = off;                                     // zero [0, zeroEnd)
  size_t o_h     = alloc((size_t)N * HEADS * COUT * sizeof(float));
  size_t o_as    = alloc((size_t)N * HEADS * sizeof(float));
  size_t o_ad    = alloc((size_t)N * HEADS * sizeof(float));
  size_t o_gatH  = alloc(((size_t)N * COUT + 256) * sizeof(_Float16));
  size_t o_wh    = alloc((size_t)16 * KPAD * sizeof(_Float16));
  (void)ws_size; (void)n_in; (void)out_size;

  char* ws = (char*)d_ws;
  double*   sums  = (double*)(ws + o_sums);
  float*    denom = (float*)(ws + o_denom);
  float*    accum = (float*)(ws + o_accum);
  float*    h     = (float*)(ws + o_h);
  float*    a_s   = (float*)(ws + o_as);
  float*    a_d   = (float*)(ws + o_ad);
  _Float16* gatH  = (_Float16*)(ws + o_gatH);
  _Float16* Wh    = (_Float16*)(ws + o_wh);
  float*    out   = (float*)d_out;

  // 1) zero atomic accumulators (sums + denom + accum) and gatH padding tail
  long long zf = (long long)(zeroEnd / sizeof(float));
  k_zero<<<(unsigned)((zf + 255) / 256), 256, 0, stream>>>((float*)ws, zf);
  k_zero<<<1, 128, 0, stream>>>((float*)(ws + o_gatH + (size_t)N * COUT * sizeof(_Float16)), 128);

  // 2) BN statistics
  k_bnstats<<<1024, 256, 0, stream>>>(x, sums, N);

  // 3) BN apply + linear + attention coefficients
  {
    long long t = (long long)N * HEADS;
    k_linear<<<(unsigned)((t + 255) / 256), 256, 0, stream>>>(x, sums, gamma, beta, lin_w,
                                                              att_src, att_dst, h, a_s, a_d, N);
  }

  // 4) softmax denominators over edges + self loops, then aggregation
  {
    long long t = (long long)E + N;
    k_edge_denom<<<(unsigned)((t + 255) / 256), 256, 0, stream>>>(ei, a_s, a_d, denom, E, N);
    k_edge_aggr<<<(unsigned)((t + 255) / 256), 256, 0, stream>>>(ei, a_s, a_d, denom, h, accum, E, N);
  }

  // 5) bias + ELU -> f16 activations
  {
    long long t = (long long)N * COUT;
    k_elu<<<(unsigned)((t + 255) / 256), 256, 0, stream>>>(accum, gbias, gatH, t);
  }

  // 6) conv weight rearrangement (f16, zero-padded to KPAD)
  k_prep_w<<<(16 * KPAD + 255) / 256, 256, 0, stream>>>(conv_w, Wh);

  // 7) Conv1d via WMMA: B blocks, 8 waves/block, one 16x16 tile per wave
  k_conv<<<B, 256, 0, stream>>>(gatH, Wh, conv_b, out);
}
